// SpatialAttention_9981503996547
// MI455X (gfx1250) — matrix operand order repaired and checked by reading, not compiled, not run on hardware
//
#include <hip/hip_runtime.h>

#define B_   4
#define C_   256
#define CQK_ 64
#define N_   4096
#define CV_  (C_ + 16)   // V channels + one "ones" tile for the softmax denominator

#define LOG2E_ 1.44269504088896340736f

typedef __attribute__((ext_vector_type(16))) __bf16 v16bf;
typedef __attribute__((ext_vector_type(8)))  float  v8f;

union Frag16 {
  v16bf v;
  uint4 q[2];
  unsigned short s[16];
};

// Native f32 -> bf16 convert (RNE on gfx1250); backend may pack pairs.
__device__ __forceinline__ unsigned short f2bf(float f) {
  union { __bf16 h; unsigned short s; } u;
  u.h = (__bf16)f;
  return u.s;
}

__device__ __forceinline__ v8f wmma_bf16(const Frag16& a, const Frag16& b, v8f c) {
  return __builtin_amdgcn_wmma_f32_16x16x32_bf16(false, a.v, false, b.v,
                                                 (short)0, c, false, false);
}

// A-fragment (16xK rows of a row-major bf16 matrix). Lane l holds row (base+l%16),
// K chunks {kb+8*(l/16)..+7} and {kb+16+8*(l/16)..+7}  (ISA 7.12.2, 16-bit A 16x32).
__device__ __forceinline__ Frag16 load_a_frag(const unsigned short* M, int stride,
                                              int row, int kb, int lhalf) {
  const unsigned short* p = M + (size_t)row * stride + kb + lhalf * 8;
  Frag16 f;
  f.q[0] = *(const uint4*)(p);
  f.q[1] = *(const uint4*)(p + 16);
  return f;
}

// B-fragment (Kx16): lane l holds column (l%16), K chunks {kb+8*(l/16)..+7} and
// {kb+16+8*(l/16)..+7} of row (colrow) of the K-major-transposed source matrix.
__device__ __forceinline__ Frag16 load_b_frag(const unsigned short* M, int stride,
                                              int colrow, int kb, int lhalf) {
  const unsigned short* p = M + (size_t)colrow * stride + kb + lhalf * 8;
  Frag16 f;
  f.q[0] = *(const uint4*)(p);
  f.q[1] = *(const uint4*)(p + 16);
  return f;
}

// Compiler-only ordering for wave-private LDS traffic: DS ops from one wave are
// executed in order by hardware, so no s_barrier / waitcnt is needed, only a
// scheduling + aliasing fence.
__device__ __forceinline__ void wave_lds_fence() {
  __builtin_amdgcn_fence(__ATOMIC_ACQ_REL, "wavefront");
  __builtin_amdgcn_wave_barrier();
}

// ---------------- phase 0: dtype conversion / transpose / fill ----------------

__global__ void cvt_f32_bf16_kernel(const float* __restrict__ src,
                                    unsigned short* __restrict__ dst, int n) {
  int i = blockIdx.x * blockDim.x + threadIdx.x;
  if (i < n) dst[i] = f2bf(src[i]);
}

// x[B,C,N] fp32 -> xt[B,N,C] bf16, LDS-tiled so both sides are coalesced.
__global__ void transpose_cvt_kernel(const float* __restrict__ x,
                                     unsigned short* __restrict__ xt) {
  __shared__ float tile[16][17];
  const int b  = blockIdx.z;
  const int n0 = blockIdx.x * 16;
  const int c0 = blockIdx.y * 16;
  const int tx = threadIdx.x & 15;
  const int ty = threadIdx.x >> 4;
  tile[ty][tx] = x[((size_t)b * C_ + c0 + ty) * N_ + n0 + tx];
  __syncthreads();
  xt[((size_t)b * N_ + n0 + ty) * C_ + c0 + tx] = f2bf(tile[tx][ty]);
}

// V channels 256..271: channel 256 = 1.0 (denominator column), 257..271 = 0.
__global__ void fill_ones_kernel(unsigned short* __restrict__ vC) {
  size_t i = (size_t)blockIdx.x * blockDim.x + threadIdx.x;
  const size_t total = (size_t)B_ * 16 * N_;
  if (i >= total) return;
  const int b    = (int)(i / (16 * N_));
  const int rem  = (int)(i % (16 * N_));
  const int crow = rem / N_;   // 0..15 -> channel 256+crow
  const int n    = rem % N_;
  vC[((size_t)b * CV_ + C_ + crow) * N_ + n] =
      (crow == 0) ? (unsigned short)0x3F80 : (unsigned short)0;
}

// ---------------- phase 1: projections (WMMA GEMMs) ----------------

// q = (Wq x + bq) * log2(e)  [exp2-domain softmax], k = Wk x + bk.
// Outputs stored transposed: qT/kT[B,N,CQK] bf16.
__global__ __launch_bounds__(128, 1)
void proj_qk_kernel(const unsigned short* __restrict__ xt,
                    const unsigned short* __restrict__ Wqb,
                    const unsigned short* __restrict__ Wkb,
                    const float* __restrict__ bq,
                    const float* __restrict__ bk,
                    unsigned short* __restrict__ qT,
                    unsigned short* __restrict__ kT) {
  const int b     = blockIdx.y;
  const int wave  = threadIdx.x >> 5;
  const int lane  = threadIdx.x & 31;
  const int lhalf = lane >> 4;
  const int lmod  = lane & 15;
  const int n0    = (blockIdx.x * 4 + wave) << 4;

  const unsigned short* xtb = xt + (size_t)b * N_ * C_;
  const v8f vzero = {0.f, 0.f, 0.f, 0.f, 0.f, 0.f, 0.f, 0.f};
  v8f aq[4], ak[4];
#pragma unroll
  for (int t = 0; t < 4; ++t) { aq[t] = vzero; ak[t] = vzero; }

  for (int kb = 0; kb < C_; kb += 32) {
    Frag16 xf = load_b_frag(xtb, C_, n0 + lmod, kb, lhalf);
#pragma unroll
    for (int ot = 0; ot < 4; ++ot) {
      Frag16 wq = load_a_frag(Wqb, C_, ot * 16 + lmod, kb, lhalf);
      aq[ot] = wmma_bf16(wq, xf, aq[ot]);
      Frag16 wk = load_a_frag(Wkb, C_, ot * 16 + lmod, kb, lhalf);
      ak[ot] = wmma_bf16(wk, xf, ak[ot]);
    }
  }

  unsigned short* qrow = qT + (size_t)b * N_ * CQK_ + (size_t)(n0 + lmod) * CQK_;
  unsigned short* krow = kT + (size_t)b * N_ * CQK_ + (size_t)(n0 + lmod) * CQK_;
#pragma unroll
  for (int ot = 0; ot < 4; ++ot) {
    const int obase = ot * 16 + 8 * lhalf;  // lane's 8 consecutive output rows o
    float4 q0 = *(const float4*)(bq + obase);
    float4 q1 = *(const float4*)(bq + obase + 4);
    float4 k0 = *(const float4*)(bk + obase);
    float4 k1 = *(const float4*)(bk + obase + 4);
    uint4 pk;
    pk.x = f2bf((aq[ot][0] + q0.x) * LOG2E_) |
           ((unsigned)f2bf((aq[ot][1] + q0.y) * LOG2E_) << 16);
    pk.y = f2bf((aq[ot][2] + q0.z) * LOG2E_) |
           ((unsigned)f2bf((aq[ot][3] + q0.w) * LOG2E_) << 16);
    pk.z = f2bf((aq[ot][4] + q1.x) * LOG2E_) |
           ((unsigned)f2bf((aq[ot][5] + q1.y) * LOG2E_) << 16);
    pk.w = f2bf((aq[ot][6] + q1.z) * LOG2E_) |
           ((unsigned)f2bf((aq[ot][7] + q1.w) * LOG2E_) << 16);
    *(uint4*)(qrow + obase) = pk;
    pk.x = f2bf(ak[ot][0] + k0.x) | ((unsigned)f2bf(ak[ot][1] + k0.y) << 16);
    pk.y = f2bf(ak[ot][2] + k0.z) | ((unsigned)f2bf(ak[ot][3] + k0.w) << 16);
    pk.z = f2bf(ak[ot][4] + k1.x) | ((unsigned)f2bf(ak[ot][5] + k1.y) << 16);
    pk.w = f2bf(ak[ot][6] + k1.z) | ((unsigned)f2bf(ak[ot][7] + k1.w) << 16);
    *(uint4*)(krow + obase) = pk;
  }
}

// v = Wv x + bv ; stored v[B,CV,N] bf16 (c-major: PV B-fragment wants K=j contiguous).
__global__ __launch_bounds__(128, 1)
void proj_v_kernel(const unsigned short* __restrict__ xt,
                   const unsigned short* __restrict__ Wvb,
                   const float* __restrict__ bv,
                   unsigned short* __restrict__ vC) {
  const int b     = blockIdx.y;
  const int wave  = threadIdx.x >> 5;
  const int lane  = threadIdx.x & 31;
  const int lhalf = lane >> 4;
  const int lmod  = lane & 15;
  const int n0    = (blockIdx.x * 4 + wave) << 4;

  const unsigned short* xtb = xt + (size_t)b * N_ * C_;
  const v8f vzero = {0.f, 0.f, 0.f, 0.f, 0.f, 0.f, 0.f, 0.f};
  v8f acc[16];
#pragma unroll
  for (int t = 0; t < 16; ++t) acc[t] = vzero;

  for (int kb = 0; kb < C_; kb += 32) {
    Frag16 xf = load_b_frag(xtb, C_, n0 + lmod, kb, lhalf);
#pragma unroll
    for (int ot = 0; ot < 16; ++ot) {
      Frag16 wv = load_a_frag(Wvb, C_, ot * 16 + lmod, kb, lhalf);
      acc[ot] = wmma_bf16(wv, xf, acc[ot]);
    }
  }

  unsigned short* vbp = vC + (size_t)b * CV_ * N_;
  const int n = n0 + lmod;
#pragma unroll
  for (int ot = 0; ot < 16; ++ot) {
#pragma unroll
    for (int r = 0; r < 8; ++r) {
      const int c = ot * 16 + 8 * lhalf + r;
      vbp[(size_t)c * N_ + n] = f2bf(acc[ot][r] + bv[c]);
    }
  }
}

// ---------------- phase 2: flash attention + residual ----------------

__global__ __launch_bounds__(128, 1)
void attn_kernel(const unsigned short* __restrict__ qT,
                 const unsigned short* __restrict__ kT,
                 const unsigned short* __restrict__ vC,
                 const float* __restrict__ x,
                 const float* __restrict__ wscale,
                 float* __restrict__ out) {
  const int b     = blockIdx.y;
  const int wave  = threadIdx.x >> 5;
  const int lane  = threadIdx.x & 31;
  const int lhalf = lane >> 4;
  const int lmod  = lane & 15;
  const int i0    = (blockIdx.x * 4 + wave) << 4;  // 16 query rows per wave

  const unsigned short* qTb = qT + (size_t)b * N_ * CQK_;
  const unsigned short* kTb = kT + (size_t)b * N_ * CQK_;
  const unsigned short* vb  = vC + (size_t)b * CV_ * N_;

  // Per-wave P staging buffer (wave-private: no inter-wave sync required).
  __shared__ __align__(16) __bf16 plds[4][16][32];

  // Q A-fragments for K = 0..31 and 32..63 (held for the whole loop)
  Frag16 qa0 = load_a_frag(qTb, CQK_, i0 + lmod, 0, lhalf);
  Frag16 qa1 = load_a_frag(qTb, CQK_, i0 + lmod, 32, lhalf);

  const v8f vzero = {0.f, 0.f, 0.f, 0.f, 0.f, 0.f, 0.f, 0.f};
  // 16 rows x 272 channels fp32 accumulator; tile 16 column 0 carries the
  // softmax denominator l (V "ones" column) through the same flash recurrence.
  v8f o[17];
#pragma unroll
  for (int t = 0; t < 17; ++t) o[t] = vzero;
  float m_[8];
#pragma unroll
  for (int r = 0; r < 8; ++r) m_[r] = -3.0e38f;

  for (int j0 = 0; j0 < N_; j0 += 32) {
    // ---- S = Qᵀ·K for 16(i) x 32(j)  (q pre-scaled by log2e) ----
    v8f s0 = vzero, s1 = vzero;
    {
      Frag16 kf0 = load_b_frag(kTb, CQK_, j0 + lmod, 0, lhalf);
      Frag16 kf1 = load_b_frag(kTb, CQK_, j0 + lmod, 32, lhalf);
      s0 = wmma_bf16(qa0, kf0, s0);
      s0 = wmma_bf16(qa1, kf1, s0);
      Frag16 kf2 = load_b_frag(kTb, CQK_, j0 + 16 + lmod, 0, lhalf);
      Frag16 kf3 = load_b_frag(kTb, CQK_, j0 + 16 + lmod, 32, lhalf);
      s1 = wmma_bf16(qa0, kf2, s1);
      s1 = wmma_bf16(qa1, kf3, s1);
    }
    if (j0 + 32 < N_)  // global_prefetch_b8 of the next K tile
      __builtin_prefetch(kTb + (size_t)(j0 + 32 + lmod) * CQK_, 0, 0);

    // ---- online softmax (base 2): row max over this 32-key chunk ----
    // C-layout: VGPR r -> row r (lanes 0-15) / row r+8 (lanes 16-31), col = lane%16,
    // so a row reduction is a 16-lane xor-butterfly.
    float mx[8];
#pragma unroll
    for (int r = 0; r < 8; ++r) mx[r] = fmaxf(s0[r], s1[r]);
#pragma unroll
    for (int d = 1; d < 16; d <<= 1)
#pragma unroll
      for (int r = 0; r < 8; ++r)
        mx[r] = fmaxf(mx[r], __shfl_xor(mx[r], d, 32));

    // Rescale all accumulators (incl. the denominator tile) only when some
    // row's running max grew (wave-uniform vote; factors are exactly 1 otherwise).
    bool grew = false;
#pragma unroll
    for (int r = 0; r < 8; ++r) grew = grew || (mx[r] > m_[r]);
    if (__any(grew)) {
      float scale[8];
#pragma unroll
      for (int r = 0; r < 8; ++r) {
        float mn = fmaxf(m_[r], mx[r]);
        scale[r] = exp2f(m_[r] - mn);
        m_[r] = mn;
      }
#pragma unroll
      for (int t = 0; t < 17; ++t)
#pragma unroll
        for (int r = 0; r < 8; ++r)
          o[t][r] *= scale[r];
    }

    // ---- P = exp2(S - m), stage P to LDS in [i][j] order ----
#pragma unroll
    for (int r = 0; r < 8; ++r) {
      float p0 = exp2f(s0[r] - m_[r]);
      float p1 = exp2f(s1[r] - m_[r]);
      plds[wave][r + 8 * lhalf][lmod]      = (__bf16)p0;
      plds[wave][r + 8 * lhalf][16 + lmod] = (__bf16)p1;
    }

    wave_lds_fence();  // stores above visible to the transposed reload below
    Frag16 pa;
    {
      const unsigned short* prow = (const unsigned short*)&plds[wave][lmod][0];
      pa.q[0] = *(const uint4*)(prow + lhalf * 8);
      pa.q[1] = *(const uint4*)(prow + 16 + lhalf * 8);
    }
    wave_lds_fence();  // reads done before next iteration overwrites plds

    // ---- O += P·Vᵀ over 16 channel tiles + denominator tile ----
#pragma unroll
    for (int ct = 0; ct < 17; ++ct) {
      Frag16 vf = load_b_frag(vb, N_, ct * 16 + lmod, j0, lhalf);
      o[ct] = wmma_bf16(pa, vf, o[ct]);
    }
  }

  // ---- epilogue: out = weight * O / l + x ----
  // l for rows 0..7 sits in lane 0 (c=256 column), rows 8..15 in lane 16.
  const float wgt = wscale[0];
  const float* xb = x + (size_t)b * C_ * N_;
  float* ob = out + (size_t)b * C_ * N_;
  float invl[8];
#pragma unroll
  for (int r = 0; r < 8; ++r) {
    float lsum = __shfl(o[16][r], lane & 16, 32);
    invl[r] = 1.0f / lsum;
  }
  const int ibase = i0 + 8 * lhalf;  // lane's 8 consecutive query rows
#pragma unroll
  for (int ct = 0; ct < 16; ++ct) {
    const int c = ct * 16 + lmod;
    const float* xp = xb + (size_t)c * N_ + ibase;
    float* op = ob + (size_t)c * N_ + ibase;
    float4 x0 = *(const float4*)(xp);
    float4 x1 = *(const float4*)(xp + 4);
    float4 r0, r1;
    r0.x = wgt * o[ct][0] * invl[0] + x0.x;
    r0.y = wgt * o[ct][1] * invl[1] + x0.y;
    r0.z = wgt * o[ct][2] * invl[2] + x0.z;
    r0.w = wgt * o[ct][3] * invl[3] + x0.w;
    r1.x = wgt * o[ct][4] * invl[4] + x1.x;
    r1.y = wgt * o[ct][5] * invl[5] + x1.y;
    r1.z = wgt * o[ct][6] * invl[6] + x1.z;
    r1.w = wgt * o[ct][7] * invl[7] + x1.w;
    *(float4*)(op) = r0;
    *(float4*)(op + 4) = r1;
  }
}

// ---------------- host launch ----------------

extern "C" void kernel_launch(void* const* d_in, const int* in_sizes, int n_in,
                              void* d_out, int out_size, void* d_ws, size_t ws_size,
                              hipStream_t stream) {
  const float* features = (const float*)d_in[0];
  const float* Wq = (const float*)d_in[1];
  const float* bq = (const float*)d_in[2];
  const float* Wk = (const float*)d_in[3];
  const float* bk = (const float*)d_in[4];
  const float* Wv = (const float*)d_in[5];
  const float* bv = (const float*)d_in[6];
  const float* wt = (const float*)d_in[7];
  float* out = (float*)d_out;

  char* ws = (char*)d_ws;
  unsigned short* xt  = (unsigned short*)ws; ws += (size_t)B_ * N_ * C_ * 2;
  unsigned short* qT  = (unsigned short*)ws; ws += (size_t)B_ * N_ * CQK_ * 2;
  unsigned short* kT  = (unsigned short*)ws; ws += (size_t)B_ * N_ * CQK_ * 2;
  unsigned short* vC  = (unsigned short*)ws; ws += (size_t)B_ * CV_ * N_ * 2;
  unsigned short* Wqb = (unsigned short*)ws; ws += (size_t)CQK_ * C_ * 2;
  unsigned short* Wkb = (unsigned short*)ws; ws += (size_t)CQK_ * C_ * 2;
  unsigned short* Wvb = (unsigned short*)ws; ws += (size_t)C_ * C_ * 2;

  cvt_f32_bf16_kernel<<<(CQK_ * C_ + 255) / 256, 256, 0, stream>>>(Wq, Wqb, CQK_ * C_);
  cvt_f32_bf16_kernel<<<(CQK_ * C_ + 255) / 256, 256, 0, stream>>>(Wk, Wkb, CQK_ * C_);
  cvt_f32_bf16_kernel<<<(C_ * C_ + 255) / 256, 256, 0, stream>>>(Wv, Wvb, C_ * C_);

  fill_ones_kernel<<<(B_ * 16 * N_ + 255) / 256, 256, 0, stream>>>(vC);

  transpose_cvt_kernel<<<dim3(N_ / 16, C_ / 16, B_), 256, 0, stream>>>(features, xt);

  proj_qk_kernel<<<dim3(N_ / 64, B_), 128, 0, stream>>>(xt, Wqb, Wkb, bq, bk, qT, kT);
  proj_v_kernel<<<dim3(N_ / 64, B_), 128, 0, stream>>>(xt, Wvb, bv, vC);

  attn_kernel<<<dim3(N_ / 64, B_), 128, 0, stream>>>(qT, kT, vC, features, wt, out);
}
